// HLALayer_62302795596550
// MI455X (gfx1250) — matrix operand order repaired and checked by reading, not compiled, not run on hardware
//
#include <hip/hip_runtime.h>
#include <hip/hip_bf16.h>
#include <cstdint>

typedef __bf16 bf16;
typedef __bf16 v16bf __attribute__((ext_vector_type(16)));
typedef float  v8f   __attribute__((ext_vector_type(8)));
typedef unsigned int u32x4 __attribute__((ext_vector_type(4)));
typedef int          i32x4 __attribute__((ext_vector_type(4)));
typedef int          i32x8 __attribute__((ext_vector_type(8)));

#define DEV static __device__ __forceinline__

// ---------------------------------------------------------------------------
// v_wmma_f32_16x16x32_bf16 wrapper
// ---------------------------------------------------------------------------
DEV v8f wmma_bf(v8f c, v16bf a, v16bf b) {
  return __builtin_amdgcn_wmma_f32_16x16x32_bf16(
      false, a, false, b, (short)0, c, false, false);
}

// ---------------------------------------------------------------------------
// Fragment loaders (CDNA5 ISA 7.12.2, wave32). Both loaders read 2x16B
// contiguous runs from LDS when the buffer is laid out along K.
//  A (16x32): lane<16 -> row M=lane, half adds 8 to K base; elems 0..7 at
//             K=kb.., elems 8..15 at K=kb+16..
//  B (32x16): col = lane&15, half adds 8 to K base; elems 0..7 at K=kb..,
//             elems 8..15 at K=kb+16.. (same as A).
//             We always source B from a [N][K]-layout ("transposed") buffer
//             so the 16 reads are contiguous.
//  C/D: elem r -> M = 8*(lane>>4)+r, N = lane&15
// ---------------------------------------------------------------------------
DEV v16bf load_a_frag(const bf16* s, int ld, int row0, int k0, int lane) {
  v16bf a = {};
  int r  = row0 + (lane & 15);
  int kb = k0 + ((lane >> 4) << 3);
#pragma unroll
  for (int j = 0; j < 8; ++j) {
    a[j]     = s[r * ld + kb + j];
    a[j + 8] = s[r * ld + kb + 16 + j];
  }
  return a;
}

// B operand from [N][K]-layout buffer: contiguous along K.
DEV v16bf load_b_fragT(const bf16* s, int ld, int k0, int col0, int lane) {
  v16bf b = {};
  int c  = col0 + (lane & 15);
  int kb = k0 + ((lane >> 4) << 3);
#pragma unroll
  for (int e = 0; e < 8; ++e) {
    b[e]     = s[c * ld + kb + e];
    b[e + 8] = s[c * ld + kb + 16 + e];
  }
  return b;
}

// 64x64x64 matmul, one 16x16 output tile (ti,tj): 2 WMMAs.
// A row-major [M][K], B in [N][K] layout.
DEV void mm64(v8f& acc, const bf16* A, const bf16* Bt, int ti, int tj, int lane) {
#pragma unroll
  for (int k0 = 0; k0 < 64; k0 += 32) {
    v16bf a = load_a_frag(A, 64, ti * 16, k0, lane);
    v16bf b = load_b_fragT(Bt, 64, k0, tj * 16, lane);
    acc = wmma_bf(acc, a, b);
  }
}

DEV void store_tile_bf(bf16* d, int i0, int j0, v8f acc, int lane, int tril_mask) {
  int jl = lane & 15, ib = (lane >> 4) << 3;
#pragma unroll
  for (int r = 0; r < 8; ++r) {
    int i = i0 + ib + r, j = j0 + jl;
    float v = acc[r];
    if (tril_mask && j > i) v = 0.f;
    d[i * 64 + j] = (bf16)v;
  }
}

DEV void store_tile_bf_T(bf16* d, int i0, int j0, v8f acc, int lane) {
  int jl = lane & 15, ib = (lane >> 4) << 3;
#pragma unroll
  for (int r = 0; r < 8; ++r) {
    int i = i0 + ib + r, j = j0 + jl;
    d[j * 64 + i] = (bf16)acc[r];
  }
}

// state += acc (f32) and refresh bf16 [N][K]-layout shadow (optionally negated)
DEV void upd_state(float* st, bf16* shadowT, int i0, int j0, v8f acc, int lane, float sgn) {
  int jl = lane & 15, ib = (lane >> 4) << 3;
#pragma unroll
  for (int r = 0; r < 8; ++r) {
    int i = i0 + ib + r, j = j0 + jl;
    float nv = st[i * 64 + j] + acc[r];
    st[i * 64 + j] = nv;
    shadowT[j * 64 + i] = (bf16)(sgn * nv);
  }
}

// ---------------------------------------------------------------------------
// TDM: 2D tile load (64 rows x 64 bf16 cols) from a row-major [*,1024] tensor
// into LDS at byte offset lds_off. D# per CDNA5 ISA.
// This toolchain exposes the 6-arg builtin:
//   (u32x4 g0, i32x8 g1, i32x4 g2, i32x4 g3, i32x8 extra, i32 cpol)
// ---------------------------------------------------------------------------
#if __has_builtin(__builtin_amdgcn_tensor_load_to_lds)
#define HAVE_TDM 1
DEV void tdm_load_tile64(const bf16* gptr, unsigned lds_off) {
  unsigned long long ga = (unsigned long long)(uintptr_t)gptr;
  u32x4 g0;
  g0[0] = 1u;                                   // count=1, user descriptor
  g0[1] = lds_off;                              // lds_addr (bytes)
  g0[2] = (unsigned)ga;                         // global_addr[31:0]
  g0[3] = (unsigned)((ga >> 32) & 0x1FFFFFFu)   // global_addr[56:32]
          | (2u << 30);                         // type=2 ("image")
  i32x8 g1;
  g1[0] = (1 << 16);          // workgroup_mask=0, data_size=1 (2 bytes)
  g1[1] = (1024 << 16);       // tensor_dim0[15:0]=1024 at bits 63:48
  g1[2] = (4096 << 16);       // tensor_dim0 hi=0; tensor_dim1[15:0]=4096
  g1[3] = (64 << 16);         // tensor_dim1 hi=0; tile_dim0=64
  g1[4] = 64;                 // tile_dim1=64; tile_dim2=0
  g1[5] = 1024;               // tensor_dim0_stride[31:0]=1024
  g1[6] = 0;                  // stride hi / tensor_dim1_stride lo
  g1[7] = 0;
  i32x4 gz4 = {0, 0, 0, 0};                       // groups 2/3 unused (2D)
  i32x8 gz8 = {0, 0, 0, 0, 0, 0, 0, 0};
  __builtin_amdgcn_tensor_load_to_lds(g0, g1, gz4, gz4, gz8, 0);
}
#else
#define HAVE_TDM 0
#endif

// ---------------------------------------------------------------------------
// f32 -> bf16
// ---------------------------------------------------------------------------
__global__ __launch_bounds__(256) void cvt_bf16(const float* __restrict__ in,
                                                bf16* __restrict__ out, int n) {
  int i = blockIdx.x * 256 + threadIdx.x;
  if (i < n) out[i] = (bf16)in[i];
}

// ---------------------------------------------------------------------------
// GEMM: C[M,N] = A[M,K] @ B[K,N]; bf16 in, f32 acc. 64x64 C tile / block,
// 8 wave32. B staged transposed in LDS so B fragments are contiguous.
// ---------------------------------------------------------------------------
template <bool OUT_F32>
__global__ __launch_bounds__(256) void gemm64(const bf16* __restrict__ A,
                                              const bf16* __restrict__ B,
                                              void* __restrict__ C,
                                              int M, int N, int K) {
  __shared__ bf16 sA[64 * 32];   // [M][K]
  __shared__ bf16 sBt[64 * 32];  // [N][K]
  int tid = threadIdx.x, lane = tid & 31, w = tid >> 5;
  int bm = blockIdx.y * 64, bn = blockIdx.x * 64;

  int t0 = 2 * w, t1 = 2 * w + 1;
  int ti0 = t0 >> 2, tj0 = t0 & 3, ti1 = t1 >> 2, tj1 = t1 & 3;
  v8f acc0 = {}, acc1 = {};

  int ra = tid >> 2, ca = (tid & 3) << 3;  // A stage: 8 bf16/thread
  int rb = tid >> 3, cb = (tid & 7) << 3;  // B stage: 8 bf16/thread

  for (int k0 = 0; k0 < K; k0 += 32) {
    *(float4*)&sA[ra * 32 + ca] = *(const float4*)&A[(size_t)(bm + ra) * K + k0 + ca];
    {
      bf16 tmp[8];
      *(float4*)tmp = *(const float4*)&B[(size_t)(k0 + rb) * N + bn + cb];
#pragma unroll
      for (int e = 0; e < 8; ++e) sBt[(cb + e) * 32 + rb] = tmp[e];
    }
    if (k0 + 32 < K) {
      __builtin_prefetch(&A[(size_t)(bm + ra) * K + k0 + 32 + ca], 0, 1);
      __builtin_prefetch(&B[(size_t)(k0 + 32 + rb) * N + bn + cb], 0, 1);
    }
    __syncthreads();
    {
      v16bf a0 = load_a_frag(sA, 32, ti0 * 16, 0, lane);
      v16bf b0 = load_b_fragT(sBt, 32, 0, tj0 * 16, lane);
      acc0 = wmma_bf(acc0, a0, b0);
      v16bf a1 = load_a_frag(sA, 32, ti1 * 16, 0, lane);
      v16bf b1 = load_b_fragT(sBt, 32, 0, tj1 * 16, lane);
      acc1 = wmma_bf(acc1, a1, b1);
    }
    __syncthreads();
  }

  int jl = lane & 15, ib = (lane >> 4) << 3;
#pragma unroll
  for (int u = 0; u < 2; ++u) {
    v8f acc = u ? acc1 : acc0;
    int ti = u ? ti1 : ti0, tj = u ? tj1 : tj0;
#pragma unroll
    for (int r = 0; r < 8; ++r) {
      size_t idx = (size_t)(bm + ti * 16 + ib + r) * N + (bn + tj * 16 + jl);
      if constexpr (OUT_F32) ((float*)C)[idx] = acc[r];
      else                   ((bf16*)C)[idx] = (bf16)acc[r];
    }
  }
}

// ---------------------------------------------------------------------------
// Chunked HLA scan. One block per (b,h), chunk = 64 steps, all math is
// 64x64x64 WMMA:
//   A = Q K^T ; P = tril(A) ; R = stril(A^T)
//   U = Q S0 ; Tm = K C0 + R V
//   W = tril(A P^T + U Q^T)
//   O = W V + U C0 - Q G0
//   S0 += K^T K ; C0 += Q^T V ; G0 += K^T Tm
// f32 state in LDS + bf16 [N][K]-layout shadows (Gbt pre-negated).
// ---------------------------------------------------------------------------
// LDS byte offsets (dynamic segment, no static LDS in this kernel)
#define L_SQ   0u        // [t][dk] row-major, TDM dest
#define L_SK   8192u     // [t][dk] row-major, TDM dest
#define L_SW   16384u    // V stage (TDM dest) -> later R / W
#define L_SQT  24576u    // [dk][t]
#define L_SKT  32768u    // [dk][t]
#define L_SVT  40960u    // [dv][t]
#define L_SAB  49152u    // A
#define L_SP   57344u    // P = tril(A)
#define L_SU   65536u    // U
#define L_STT  73728u    // Tm transposed [n][k]
#define L_SBT  81920u    // S0 shadow [n][k]
#define L_CBT  90112u    // C0 shadow [n][k]
#define L_GBT  98304u    // -G0 shadow [n][k]
#define L_SST  106496u   // f32 S0
#define L_CST  122880u   // f32 C0
#define L_GST  139264u   // f32 G0
#define SCAN_SMEM 155648u

__global__ __launch_bounds__(256) void hla_scan(const bf16* __restrict__ Qg,
                                                const bf16* __restrict__ Kg,
                                                const bf16* __restrict__ Vg,
                                                bf16* __restrict__ Og) {
  extern __shared__ char smem[];
  bf16* sQ  = (bf16*)(smem + L_SQ);
  bf16* sK  = (bf16*)(smem + L_SK);
  bf16* sW  = (bf16*)(smem + L_SW);
  bf16* sQt = (bf16*)(smem + L_SQT);
  bf16* sKt = (bf16*)(smem + L_SKT);
  bf16* sVt = (bf16*)(smem + L_SVT);
  bf16* sA  = (bf16*)(smem + L_SAB);
  bf16* sP  = (bf16*)(smem + L_SP);
  bf16* sU  = (bf16*)(smem + L_SU);
  bf16* sTt = (bf16*)(smem + L_STT);
  bf16* Sbt = (bf16*)(smem + L_SBT);
  bf16* Cbt = (bf16*)(smem + L_CBT);
  bf16* Gbt = (bf16*)(smem + L_GBT);
  float* Sst = (float*)(smem + L_SST);
  float* Cst = (float*)(smem + L_CST);
  float* Gst = (float*)(smem + L_GST);

  int tid = threadIdx.x, lane = tid & 31, w = tid >> 5;
  int bh = blockIdx.x;
  int b = bh >> 4, h = bh & 15;
  const int LD = 1024;
  size_t base = (size_t)b * 2048 * LD + (size_t)h * 64;

  for (int i = tid; i < 4096; i += 256) {
    Sst[i] = 0.f; Cst[i] = 0.f; Gst[i] = 0.f;
    Sbt[i] = (bf16)0.f; Cbt[i] = (bf16)0.f; Gbt[i] = (bf16)0.f;
  }
  __syncthreads();

  int t0 = 2 * w, t1 = 2 * w + 1;
  int TI[2] = {t0 >> 2, t1 >> 2};
  int TJ[2] = {t0 & 3, t1 & 3};
  int lr = tid >> 2, lc = (tid & 3) << 4;   // stage/scatter: 16 elems/thread

#pragma unroll 1
  for (int ch = 0; ch < 32; ++ch) {
    size_t rbase = base + (size_t)ch * 64 * LD;

    // ---- stage Q,K (row-major) and V (into sW) ----
#if HAVE_TDM
    if (w == 0) {   // TDM ignores EXEC; issue once per workgroup (wave 0)
      tdm_load_tile64(Qg + rbase, L_SQ);
      tdm_load_tile64(Kg + rbase, L_SK);
      tdm_load_tile64(Vg + rbase, L_SW);
      __builtin_amdgcn_s_wait_tensorcnt(0);
    }
#else
    {
      size_t g = rbase + (size_t)lr * LD + lc;
      *(float4*)&sQ[lr * 64 + lc]     = *(const float4*)&Qg[g];
      *(float4*)&sQ[lr * 64 + lc + 8] = *(const float4*)&Qg[g + 8];
      *(float4*)&sK[lr * 64 + lc]     = *(const float4*)&Kg[g];
      *(float4*)&sK[lr * 64 + lc + 8] = *(const float4*)&Kg[g + 8];
      *(float4*)&sW[lr * 64 + lc]     = *(const float4*)&Vg[g];
      *(float4*)&sW[lr * 64 + lc + 8] = *(const float4*)&Vg[g + 8];
    }
#endif
    __syncthreads();

    // ---- build transposed copies sQt/sKt/sVt ----
    {
      bf16 tq[16], tk[16], tv[16];
      *(float4*)&tq[0] = *(const float4*)&sQ[lr * 64 + lc];
      *(float4*)&tq[8] = *(const float4*)&sQ[lr * 64 + lc + 8];
      *(float4*)&tk[0] = *(const float4*)&sK[lr * 64 + lc];
      *(float4*)&tk[8] = *(const float4*)&sK[lr * 64 + lc + 8];
      *(float4*)&tv[0] = *(const float4*)&sW[lr * 64 + lc];
      *(float4*)&tv[8] = *(const float4*)&sW[lr * 64 + lc + 8];
#pragma unroll
      for (int e = 0; e < 16; ++e) {
        sQt[(lc + e) * 64 + lr] = tq[e];
        sKt[(lc + e) * 64 + lr] = tk[e];
        sVt[(lc + e) * 64 + lr] = tv[e];
      }
    }
    __syncthreads();

    // ---- Phase A: A = Q K^T -> sA, sP = tril(A), sW = R = stril(A^T) ----
#pragma unroll
    for (int u = 0; u < 2; ++u) {
      v8f acc = {};
      mm64(acc, sQ, sK, TI[u], TJ[u], lane);   // B=K^T: sK rows along K
      int jl = lane & 15, ib = (lane >> 4) << 3;
      int i0 = TI[u] * 16, j0 = TJ[u] * 16;
#pragma unroll
      for (int r = 0; r < 8; ++r) {
        int i = i0 + ib + r, j = j0 + jl;
        bf16 vb = (bf16)acc[r];
        sA[i * 64 + j] = vb;
        sP[i * 64 + j] = (j <= i) ? vb : (bf16)0.f;
        sW[j * 64 + i] = (i < j) ? vb : (bf16)0.f;   // R[s,r]=A[r,s], r<s
      }
    }
    __syncthreads();

    // ---- Phase B: U = Q S0 -> sU ; Tm = K C0 + R V -> sTt (transposed) ----
#pragma unroll
    for (int u = 0; u < 2; ++u) {
      v8f acc = {};
      mm64(acc, sQ, Sbt, TI[u], TJ[u], lane);
      store_tile_bf(sU, TI[u] * 16, TJ[u] * 16, acc, lane, 0);
      v8f acc2 = {};
      mm64(acc2, sK, Cbt, TI[u], TJ[u], lane);
      mm64(acc2, sW, sVt, TI[u], TJ[u], lane);     // R @ V
      store_tile_bf_T(sTt, TI[u] * 16, TJ[u] * 16, acc2, lane);
    }
    __syncthreads();

    // ---- Phase C: W = tril(A P^T + U Q^T) -> sW ----
#pragma unroll
    for (int u = 0; u < 2; ++u) {
      v8f acc = {};
      mm64(acc, sA, sP, TI[u], TJ[u], lane);       // B=P^T: sP rows along K
      mm64(acc, sU, sQ, TI[u], TJ[u], lane);       // B=Q^T: sQ rows along K
      store_tile_bf(sW, TI[u] * 16, TJ[u] * 16, acc, lane, 1);
    }
    __syncthreads();

    // ---- Phase D: O = W V + U C0 - Q G0 -> global (Gbt pre-negated) ----
#pragma unroll
    for (int u = 0; u < 2; ++u) {
      v8f acc = {};
      mm64(acc, sW, sVt, TI[u], TJ[u], lane);
      mm64(acc, sU, Cbt, TI[u], TJ[u], lane);
      mm64(acc, sQ, Gbt, TI[u], TJ[u], lane);
      int jl = lane & 15, ib = (lane >> 4) << 3;
      int i0 = TI[u] * 16, j0 = TJ[u] * 16;
#pragma unroll
      for (int r = 0; r < 8; ++r)
        Og[rbase + (size_t)(i0 + ib + r) * LD + j0 + jl] = (bf16)acc[r];
    }
    __syncthreads();

    // ---- Phase E: S0 += K^T K ; C0 += Q^T V ; G0 += K^T Tm ----
#pragma unroll
    for (int u = 0; u < 2; ++u) {
      v8f aS = {};
      mm64(aS, sKt, sKt, TI[u], TJ[u], lane);      // A=K^T, B=K
      upd_state(Sst, Sbt, TI[u] * 16, TJ[u] * 16, aS, lane, 1.f);
      v8f aC = {};
      mm64(aC, sQt, sVt, TI[u], TJ[u], lane);      // A=Q^T, B=V
      upd_state(Cst, Cbt, TI[u] * 16, TJ[u] * 16, aC, lane, 1.f);
      v8f aG = {};
      mm64(aG, sKt, sTt, TI[u], TJ[u], lane);      // A=K^T, B=Tm
      upd_state(Gst, Gbt, TI[u] * 16, TJ[u] * 16, aG, lane, -1.f);
    }
    __syncthreads();
  }
}

// ---------------------------------------------------------------------------
// Host side
// ---------------------------------------------------------------------------
extern "C" void kernel_launch(void* const* d_in, const int* in_sizes, int n_in,
                              void* d_out, int out_size, void* d_ws, size_t ws_size,
                              hipStream_t stream) {
  const float* x  = (const float*)d_in[0];
  const float* Wq = (const float*)d_in[1];
  const float* Wk = (const float*)d_in[2];
  const float* Wv = (const float*)d_in[3];
  const float* Wo = (const float*)d_in[4];
  float* out = (float*)d_out;

  const size_t NX = (size_t)4096 * 1024;
  const size_t NW = (size_t)1024 * 1024;

  char* ws = (char*)d_ws;
  size_t off = 0;
  bf16* xb  = (bf16*)(ws + off); off += NX * 2;
  bf16* wqb = (bf16*)(ws + off); off += NW * 2;
  bf16* wkb = (bf16*)(ws + off); off += NW * 2;
  bf16* wvb = (bf16*)(ws + off); off += NW * 2;
  bf16* wob = (bf16*)(ws + off); off += NW * 2;
  bf16* Qb  = (bf16*)(ws + off); off += NX * 2;
  bf16* Kb  = (bf16*)(ws + off); off += NX * 2;
  bf16* Vb  = (bf16*)(ws + off); off += NX * 2;
  bf16* Ob  = (bf16*)(ws + off); off += NX * 2;

  cvt_bf16<<<(int)(NX / 256), 256, 0, stream>>>(x,  xb,  (int)NX);
  cvt_bf16<<<(int)(NW / 256), 256, 0, stream>>>(Wq, wqb, (int)NW);
  cvt_bf16<<<(int)(NW / 256), 256, 0, stream>>>(Wk, wkb, (int)NW);
  cvt_bf16<<<(int)(NW / 256), 256, 0, stream>>>(Wv, wvb, (int)NW);
  cvt_bf16<<<(int)(NW / 256), 256, 0, stream>>>(Wo, wob, (int)NW);

  dim3 g(16, 64), blk(256);
  gemm64<false><<<g, blk, 0, stream>>>(xb, wqb, (void*)Qb, 4096, 1024, 1024);
  gemm64<false><<<g, blk, 0, stream>>>(xb, wkb, (void*)Kb, 4096, 1024, 1024);
  gemm64<false><<<g, blk, 0, stream>>>(xb, wvb, (void*)Vb, 4096, 1024, 1024);

  (void)hipFuncSetAttribute(reinterpret_cast<const void*>(&hla_scan),
                            hipFuncAttributeMaxDynamicSharedMemorySize,
                            (int)SCAN_SMEM);
  hla_scan<<<32, 256, SCAN_SMEM, stream>>>(Qb, Kb, Vb, Ob);

  gemm64<true><<<g, blk, 0, stream>>>(Ob, wob, (void*)out, 4096, 1024, 1024);
}
